// PNA_19404662243723
// MI455X (gfx1250) — matrix operand order repaired and checked by reading, not compiled, not run on hardware
//
#include <hip/hip_runtime.h>
#include <hip/hip_bf16.h>

typedef __bf16 bf16;
typedef __attribute__((ext_vector_type(16))) __bf16 v16bf;
typedef __attribute__((ext_vector_type(8)))  __bf16 v8bf;
typedef __attribute__((ext_vector_type(8)))  float  v8f;
typedef int v4i_vs __attribute__((__vector_size__(16)));

#define FDIM 64
#define NEG_INF_KEY 0x007fffffu  /* f2key(-inf) */

#define WMMA_BF16(a, b, c) \
  __builtin_amdgcn_wmma_f32_16x16x32_bf16(false, (a), false, (b), (short)0, (c), false, false)

#if __has_builtin(__builtin_amdgcn_global_load_async_to_lds_b128)
#define HAVE_ASYNC_LDS 1
#else
#define HAVE_ASYNC_LDS 0
#endif

__device__ __forceinline__ void async_copy16(const bf16* g, bf16* l) {
#if HAVE_ASYNC_LDS
  __builtin_amdgcn_global_load_async_to_lds_b128(
      (__attribute__((address_space(1))) v4i_vs*)g,
      (__attribute__((address_space(3))) v4i_vs*)l, 0, 0);
#else
  *(v8bf*)l = *(const v8bf*)g;
#endif
}

__device__ __forceinline__ void wait_gather() {
#if HAVE_ASYNC_LDS
#if __has_builtin(__builtin_amdgcn_s_wait_asynccnt)
  __builtin_amdgcn_s_wait_asynccnt(0);
#else
  asm volatile("s_wait_asynccnt 0" ::: "memory");
#endif
#endif
  asm volatile("s_wait_dscnt 0" ::: "memory");
}

// ---- monotone float <-> uint key for atomic max over floats ----
__device__ __forceinline__ unsigned int f2key(float f) {
  unsigned int u = __float_as_uint(f);
  return (u & 0x80000000u) ? ~u : (u | 0x80000000u);
}
__device__ __forceinline__ float key2f(unsigned int k) {
  return (k & 0x80000000u) ? __uint_as_float(k & 0x7fffffffu)
                           : __uint_as_float(~k);
}

// ---- WMMA fragment loads from LDS (row-major tiles, bf16) ----
// A (16xK tile, M rows): lane holds row M=lane&15; K(e) = (e/8)*16 + (lane/16)*8 + e%8
__device__ __forceinline__ v16bf frag_A(const bf16* base, int ldk) {
  int lane = threadIdx.x & 31;
  const bf16* p = base + (lane & 15) * ldk + ((lane >> 4) << 3);
  v8bf lo = *(const v8bf*)(p);
  v8bf hi = *(const v8bf*)(p + 16);
  v16bf f;
#pragma unroll
  for (int i = 0; i < 8; ++i) { f[i] = lo[i]; f[i + 8] = hi[i]; }
  return f;
}
// B (Kx16 tile stored as [n][K]): lane holds col N=lane&15; K(e) = (e/8)*16 + (lane/16)*8 + e%8
__device__ __forceinline__ v16bf frag_B(const bf16* base, int ldk) {
  int lane = threadIdx.x & 31;
  const bf16* p = base + (lane & 15) * ldk + ((lane >> 4) << 3);
  v8bf lo = *(const v8bf*)(p);
  v8bf hi = *(const v8bf*)(p + 16);
  v16bf f;
#pragma unroll
  for (int i = 0; i < 8; ++i) { f[i] = lo[i]; f[i + 8] = hi[i]; }
  return f;
}

// =================== small utility kernels ===================
__global__ void pna_f2bf(const float* __restrict__ in, bf16* __restrict__ out, int n) {
  int i = blockIdx.x * blockDim.x + threadIdx.x;
  if (i < n) out[i] = (bf16)in[i];
}
__global__ void pna_init_agg(float* __restrict__ sum, unsigned int* __restrict__ mkey, int n) {
  int i = blockIdx.x * blockDim.x + threadIdx.x;
  if (i < n) { sum[i] = 0.f; mkey[i] = NEG_INF_KEY; }
}
__global__ void pna_init_cnt(float* __restrict__ cnt, int n) {
  int i = blockIdx.x * blockDim.x + threadIdx.x;
  if (i < n) cnt[i] = 0.f;
}
__global__ void pna_count(const long long* __restrict__ dst, float* __restrict__ cnt, int nE) {
  int i = blockIdx.x * blockDim.x + threadIdx.x;
  if (i < nE) atomicAdd(&cnt[dst[i]], 1.0f);
}

// =================== edge message + scatter-aggregate ===================
// m_e = Wpre * [x_dst ; x_src] + bpre ; scatter sum/max to dst node
#define EK_WAVES 8
__global__ __launch_bounds__(EK_WAVES * 32)
void pna_edge_msg(const bf16* __restrict__ xb,          // [N][64] bf16
                  const long long* __restrict__ src,
                  const long long* __restrict__ dst,
                  const float* __restrict__ Wpre,       // [64][128] row-major
                  const float* __restrict__ bpre,       // [64]
                  float* __restrict__ sumO,             // [N][64]
                  unsigned int* __restrict__ mkeyO,     // [N][64]
                  int nE, int nTiles) {
  __shared__ __align__(16) bf16 wlds[64 * 128];         // 16 KB
  __shared__ float blds[64];
  __shared__ __align__(16) bf16 featAll[EK_WAVES][16 * 128]; // 32 KB

  int tid = threadIdx.x;
  for (int i = tid; i < 64 * 128; i += blockDim.x) wlds[i] = (bf16)Wpre[i];
  if (tid < 64) blds[tid] = bpre[tid];
  __syncthreads();

  int lane = tid & 31, wave = tid >> 5;
  bf16* feat = featAll[wave];
  int waveG = blockIdx.x * EK_WAVES + wave;
  int waveN = gridDim.x * EK_WAVES;

  // gather mapping: 16B chunks; one edge row pair = 16 chunks = 16 lanes
  int eSub  = lane >> 4;            // which of 2 edges per issue
  int half  = (lane >> 3) & 1;      // 0: x[dst] half, 1: x[src] half
  int chunk = lane & 7;             // 16B chunk within 128B bf16 row

  for (int tile = waveG; tile < nTiles; tile += waveN) {
    int e0 = tile * 16;
    // async gather: 16 edges x 128 bf16 (= 256B/edge) into per-wave LDS tile
#pragma unroll
    for (int it = 0; it < 8; ++it) {
      int e = it * 2 + eSub;
      int edge = e0 + e;
      if (edge < nE) {
        long long row = half ? src[edge] : dst[edge];
        const bf16* g = xb + row * 64 + chunk * 8;
        bf16* l = feat + e * 128 + half * 64 + chunk * 8;
        async_copy16(g, l);
      }
    }
    wait_gather();

    int eL = e0 + (lane & 15);
    long long dn = (eL < nE) ? dst[eL] : -1;
#pragma unroll
    for (int mt = 0; mt < 4; ++mt) {
      v8f c = {};
#pragma unroll
      for (int kt = 0; kt < 4; ++kt) {
        v16bf a = frag_A(wlds + (mt * 16) * 128 + kt * 32, 128);
        v16bf b = frag_B(feat + kt * 32, 128);
        c = WMMA_BF16(a, b, c);
      }
      if (dn >= 0) {
        long long baseO = dn * 64;
        int m0 = mt * 16 + ((lane >> 4) << 3);
#pragma unroll
        for (int vg = 0; vg < 8; ++vg) {
          float val = c[vg] + blds[m0 + vg];
          atomicAdd(&sumO[baseO + m0 + vg], val);         // segment sum (L2-resident)
          atomicMax(&mkeyO[baseO + m0 + vg], f2key(val)); // segment max via key
        }
      }
    }
  }
}

// =================== node update: post + lin (+bias, relu) ===================
#define NK_WAVES 2
__global__ __launch_bounds__(NK_WAVES * 32)
void pna_node_update(const bf16* __restrict__ xb,       // [N][64] bf16
                     const float* __restrict__ sumI,    // [N][64]
                     const unsigned int* __restrict__ mkeyI,
                     const float* __restrict__ cnt,     // [N]
                     const float* __restrict__ Wpost,   // [64][256]
                     const float* __restrict__ bpost,   // [64]
                     const float* __restrict__ Wlin,    // [64][64]
                     const float* __restrict__ blin,    // [64]
                     bf16* __restrict__ outb,           // [N][64] bf16
                     int nN, int nTiles) {
  __shared__ __align__(16) bf16 wpost[64 * 256];        // 32 KB
  __shared__ __align__(16) bf16 wlin[64 * 64];          // 8 KB
  __shared__ float bp[64], bl[64];
  __shared__ __align__(16) bf16 featAll[NK_WAVES][16 * 256]; // 16 KB
  __shared__ __align__(16) bf16 hAll[NK_WAVES][16 * 64];     // 4 KB

  int tid = threadIdx.x;
  for (int i = tid; i < 64 * 256; i += blockDim.x) wpost[i] = (bf16)Wpost[i];
  for (int i = tid; i < 64 * 64; i += blockDim.x) wlin[i] = (bf16)Wlin[i];
  if (tid < 64) { bp[tid] = bpost[tid]; bl[tid] = blin[tid]; }
  __syncthreads();

  int lane = tid & 31, wave = tid >> 5;
  bf16* feat = featAll[wave];
  bf16* hl = hAll[wave];
  int waveG = blockIdx.x * NK_WAVES + wave;
  int waveN = gridDim.x * NK_WAVES;

  for (int tile = waveG; tile < nTiles; tile += waveN) {
    int n0 = tile * 16;
    int k = lane * 2;  // 2 features per lane per 64-wide segment
#pragma unroll
    for (int e = 0; e < 16; ++e) {
      int node = n0 + e;
      bf16 x0 = (bf16)0.f, x1 = (bf16)0.f;
      float s0 = 0, s1 = 0, m0 = 0, m1 = 0, inv = 1.f;
      if (node < nN) {
        long long b = (long long)node * 64;
        float cn = cnt[node];
        inv = 1.f / fmaxf(cn, 1.f);
        x0 = xb[b + k];   x1 = xb[b + k + 1];
        s0 = sumI[b + k]; s1 = sumI[b + k + 1];
        if (cn > 0.f) { m0 = key2f(mkeyI[b + k]); m1 = key2f(mkeyI[b + k + 1]); }
      }
      bf16* fp = feat + e * 256;                // [x | mean | max | sum]
      fp[k]       = x0;               fp[k + 1]       = x1;
      fp[64 + k]  = (bf16)(s0 * inv); fp[64 + k + 1]  = (bf16)(s1 * inv);
      fp[128 + k] = (bf16)m0;         fp[128 + k + 1] = (bf16)m1;
      fp[192 + k] = (bf16)s0;         fp[192 + k + 1] = (bf16)s1;
    }
    asm volatile("s_wait_dscnt 0" ::: "memory");

    // GEMM1: h[16x64] = feat[16x256] . Wpost^T + bpost
#pragma unroll
    for (int mt = 0; mt < 4; ++mt) {
      v8f c = {};
#pragma unroll
      for (int kt = 0; kt < 8; ++kt) {
        v16bf a = frag_A(wpost + (mt * 16) * 256 + kt * 32, 256);
        v16bf b = frag_B(feat + kt * 32, 256);
        c = WMMA_BF16(a, b, c);
      }
      int n = lane & 15;
      int m0 = mt * 16 + ((lane >> 4) << 3);
#pragma unroll
      for (int vg = 0; vg < 8; ++vg)
        hl[n * 64 + m0 + vg] = (bf16)(c[vg] + bp[m0 + vg]);
    }
    asm volatile("s_wait_dscnt 0" ::: "memory");

    // GEMM2: out[16x64] = relu(h . Wlin^T + blin), stored bf16
    int nodeL = n0 + (lane & 15);
#pragma unroll
    for (int mt = 0; mt < 4; ++mt) {
      v8f c = {};
#pragma unroll
      for (int kt = 0; kt < 2; ++kt) {
        v16bf a = frag_A(wlin + (mt * 16) * 64 + kt * 32, 64);
        v16bf b = frag_B(hl + kt * 32, 64);
        c = WMMA_BF16(a, b, c);
      }
      if (nodeL < nN) {
        long long b = (long long)nodeL * 64;
        int m0 = mt * 16 + ((lane >> 4) << 3);
#pragma unroll
        for (int vg = 0; vg < 8; ++vg)
          outb[b + m0 + vg] = (bf16)fmaxf(c[vg] + bl[m0 + vg], 0.f);
      }
    }
  }
}

// =================== final classifier ===================
#define CK_WAVES 8
__global__ __launch_bounds__(CK_WAVES * 32)
void pna_classifier(const bf16* __restrict__ hin,       // [N][64] bf16
                    const float* __restrict__ Wf,       // [32][64]
                    const float* __restrict__ bfv,      // [32]
                    float* __restrict__ out,            // [N][32]
                    int nN, int nTiles) {
  __shared__ __align__(16) bf16 wlds[32 * 64];          // 4 KB
  __shared__ float blds[32];
  __shared__ __align__(16) bf16 featAll[CK_WAVES][16 * 64]; // 16 KB

  int tid = threadIdx.x;
  for (int i = tid; i < 32 * 64; i += blockDim.x) wlds[i] = (bf16)Wf[i];
  if (tid < 32) blds[tid] = bfv[tid];
  __syncthreads();

  int lane = tid & 31, wave = tid >> 5;
  bf16* feat = featAll[wave];
  int waveG = blockIdx.x * CK_WAVES + wave;
  int waveN = gridDim.x * CK_WAVES;

  // staging mapping: 4 nodes per issue, 8 chunks of 16B per 128B node row
  int eSub  = lane >> 3;            // 0..3
  int chunk = lane & 7;             // 16B chunk

  for (int tile = waveG; tile < nTiles; tile += waveN) {
    int n0 = tile * 16;
#pragma unroll
    for (int it = 0; it < 4; ++it) {
      int e = it * 4 + eSub;
      int node = n0 + e;
      if (node < nN) {
        const bf16* g = hin + (long long)node * 64 + chunk * 8;
        bf16* l = feat + e * 64 + chunk * 8;
        async_copy16(g, l);
      }
    }
    wait_gather();

    int nodeL = n0 + (lane & 15);
#pragma unroll
    for (int mt = 0; mt < 2; ++mt) {
      v8f c = {};
#pragma unroll
      for (int kt = 0; kt < 2; ++kt) {
        v16bf a = frag_A(wlds + (mt * 16) * 64 + kt * 32, 64);
        v16bf b = frag_B(feat + kt * 32, 64);
        c = WMMA_BF16(a, b, c);
      }
      if (nodeL < nN) {
        long long b = (long long)nodeL * 32;
        int m0 = mt * 16 + ((lane >> 4) << 3);
#pragma unroll
        for (int vg = 0; vg < 8; ++vg)
          out[b + m0 + vg] = c[vg] + blds[m0 + vg];
      }
    }
  }
}

// =================== host orchestration ===================
extern "C" void kernel_launch(void* const* d_in, const int* in_sizes, int n_in,
                              void* d_out, int out_size, void* d_ws, size_t ws_size,
                              hipStream_t stream) {
  const float*     x    = (const float*)d_in[0];
  const long long* eidx = (const long long*)d_in[1];
  const float* W1pre  = (const float*)d_in[2];
  const float* b1pre  = (const float*)d_in[3];
  const float* W1post = (const float*)d_in[4];
  const float* b1post = (const float*)d_in[5];
  const float* W1lin  = (const float*)d_in[6];
  const float* b1lin  = (const float*)d_in[7];
  const float* W2pre  = (const float*)d_in[8];
  const float* b2pre  = (const float*)d_in[9];
  const float* W2post = (const float*)d_in[10];
  const float* b2post = (const float*)d_in[11];
  const float* W2lin  = (const float*)d_in[12];
  const float* b2lin  = (const float*)d_in[13];
  const float* Wf     = (const float*)d_in[14];
  const float* bfv    = (const float*)d_in[15];
  float* outp = (float*)d_out;

  const int nN = in_sizes[0] / FDIM;     // 100000
  const int nE = in_sizes[1] / 2;        // 1000000
  const long long* src = eidx;           // edge_index[0]
  const long long* dst = eidx + nE;      // edge_index[1]

  // workspace layout
  char* ws = (char*)d_ws;
  size_t off = 0;
  auto wsalloc = [&](size_t bytes) -> void* {
    void* p = ws + off;
    off = (off + bytes + 255) & ~(size_t)255;
    return p;
  };
  bf16*         xbB   = (bf16*)wsalloc((size_t)nN * 64 * sizeof(bf16));
  bf16*         h1b   = (bf16*)wsalloc((size_t)nN * 64 * sizeof(bf16));
  bf16*         h2b   = (bf16*)wsalloc((size_t)nN * 64 * sizeof(bf16));
  float*        sumB  = (float*)wsalloc((size_t)nN * 64 * sizeof(float));
  unsigned int* mkeyB = (unsigned int*)wsalloc((size_t)nN * 64 * sizeof(unsigned int));
  float*        cntB  = (float*)wsalloc((size_t)nN * sizeof(float));
  (void)ws_size; (void)n_in; (void)out_size;

  const int nAgg = nN * 64;
  const int eTiles = (nE + 15) / 16;
  const int nTiles = (nN + 15) / 16;

  // fp32 -> bf16 node features, degree (both shared by layers)
  pna_f2bf<<<(nAgg + 255) / 256, 256, 0, stream>>>(x, xbB, nAgg);
  pna_init_cnt<<<(nN + 255) / 256, 256, 0, stream>>>(cntB, nN);
  pna_count<<<(nE + 255) / 256, 256, 0, stream>>>(dst, cntB, nE);

  // ---- layer 1 ----
  pna_init_agg<<<(nAgg + 255) / 256, 256, 0, stream>>>(sumB, mkeyB, nAgg);
  pna_edge_msg<<<1024, EK_WAVES * 32, 0, stream>>>(xbB, src, dst, W1pre, b1pre,
                                                   sumB, mkeyB, nE, eTiles);
  pna_node_update<<<512, NK_WAVES * 32, 0, stream>>>(xbB, sumB, mkeyB, cntB,
                                                     W1post, b1post, W1lin, b1lin,
                                                     h1b, nN, nTiles);
  // ---- layer 2 ----
  pna_init_agg<<<(nAgg + 255) / 256, 256, 0, stream>>>(sumB, mkeyB, nAgg);
  pna_edge_msg<<<1024, EK_WAVES * 32, 0, stream>>>(h1b, src, dst, W2pre, b2pre,
                                                   sumB, mkeyB, nE, eTiles);
  pna_node_update<<<512, NK_WAVES * 32, 0, stream>>>(h1b, sumB, mkeyB, cntB,
                                                     W2post, b2post, W2lin, b2lin,
                                                     h2b, nN, nTiles);
  // ---- classifier ----
  pna_classifier<<<256, CK_WAVES * 32, 0, stream>>>(h2b, Wf, bfv, outp, nN, nTiles);
}
